// GraphSAGE_TG_10677288698290
// MI455X (gfx1250) — matrix operand order repaired and checked by reading, not compiled, not run on hardware
//
#include <hip/hip_runtime.h>
#include <hip/hip_bf16.h>

typedef __attribute__((ext_vector_type(16))) __bf16 v16bf;
typedef __attribute__((ext_vector_type(8)))  float  v8f;

#define DIM 128

// ---------------------------------------------------------------- utilities
__global__ void k_zero_f32(float* __restrict__ p, int n) {
    int i = blockIdx.x * blockDim.x + threadIdx.x;
    if (i < n) p[i] = 0.0f;
}

__global__ void k_f32_to_bf16(const float* __restrict__ x, __bf16* __restrict__ o, int n) {
    int i = blockIdx.x * blockDim.x + threadIdx.x;
    if (i < n) o[i] = (__bf16)x[i];
}

// degree count (mean denominator)
__global__ void k_count(const int* __restrict__ dst, float* __restrict__ cnt, int E) {
    int e = blockIdx.x * blockDim.x + threadIdx.x;
    if (e < E) atomicAdd(&cnt[dst[e]], 1.0f);
}

__global__ void k_invcnt(float* __restrict__ cnt, int n) {
    int i = blockIdx.x * blockDim.x + threadIdx.x;
    if (i < n) cnt[i] = 1.0f / fmaxf(cnt[i], 1.0f);
}

// ------------------------------------------------- edge scatter (segment sum)
// one wave per edge; lane l handles dims [4l, 4l+4). Gather bf16 (L2-resident),
// accumulate f32 via global_atomic_add_f32.
__global__ void k_scatter(const __bf16* __restrict__ h, const int* __restrict__ src,
                          const int* __restrict__ dst, float* __restrict__ agg, int E) {
    int t = blockIdx.x * blockDim.x + threadIdx.x;
    int e = t >> 5;
    int lane = t & 31;
    if (e >= E) return;
    int s = src[e];
    int d = dst[e];
    const __bf16* hp = h + (size_t)s * DIM + lane * 4;
    float* ap = agg + (size_t)d * DIM + lane * 4;
#pragma unroll
    for (int j = 0; j < 4; j++) atomicAdd(&ap[j], (float)hp[j]);
}

// ------------------------------------------------------------ fused SAGE GEMM
// out = relu?( (agg .* invcnt) @ Wl^T  +  h @ Wr^T  + bias )
// One wave owns one 16-col O-tile and sweeps FOUR 16-row M-strips, keeping the
// full bf16 B operand (both weight matrices, K=128) resident in 64 VGPRs.
// 8 WMMAs per strip -> 32 v_wmma_f32_16x16x32_bf16 per wave.
__global__ __launch_bounds__(256) void k_sage_gemm(
    const float*  __restrict__ agg,   // [N,128] f32 (pre-mean sums)
    const float*  __restrict__ invc,  // [N]     1/max(deg,1)
    const __bf16* __restrict__ h,     // [N,128] bf16
    const __bf16* __restrict__ Wl,    // [128,128] bf16 row-major (O,K)
    const __bf16* __restrict__ Wr,    // [128,128] bf16
    const float*  __restrict__ bias,  // [128]
    __bf16*       __restrict__ out,   // [N,128] bf16
    int relu, int N)
{
    const int lane = threadIdx.x & 31;
    const int wv   = threadIdx.x >> 5;     // O-tile 0..7
    const int m    = lane & 15;            // A row / B column within tile
    const int hi   = lane >> 4;            // lane half selects K sub-chunk
    const int o    = wv * 16 + m;

    // Load B once: lane half selects K = {k0+hi*8..+7, k0+16+hi*8..+7}
    v16bf bl[4], br[4];
    const __bf16* wlp = Wl + (size_t)o * DIM + hi * 8;
    const __bf16* wrp = Wr + (size_t)o * DIM + hi * 8;
#pragma unroll
    for (int ks = 0; ks < 4; ks++) {
        const int k0 = ks * 32;
#pragma unroll
        for (int i = 0; i < 16; i++) {
            bl[ks][i] = wlp[k0 + (i < 8 ? i : i + 8)];
            br[ks][i] = wrp[k0 + (i < 8 ? i : i + 8)];
        }
    }
    const float bb = bias[o];

#pragma unroll
    for (int t = 0; t < 4; t++) {
        const int strip = blockIdx.x * 4 + t;
        if (strip * 16 >= N) break;
        const int node = strip * 16 + m;
        const float   inv = invc[node];
        const float*  ap  = agg + (size_t)node * DIM + hi * 8;
        const __bf16* hp  = h   + (size_t)node * DIM + hi * 8;

        v8f acc = {};
#pragma unroll
        for (int ks = 0; ks < 4; ks++) {
            const int k0 = ks * 32;
            v16bf a1, a2;
            // A operand: lanes 0-15 hold K = {k0..k0+7, k0+16..k0+23},
            //            lanes 16-31 hold K = {k0+8..k0+15, k0+24..k0+31}
#pragma unroll
            for (int i = 0; i < 8; i++) {
                a1[i]     = (__bf16)(ap[k0 + i] * inv);        // mean fused here
                a1[i + 8] = (__bf16)(ap[k0 + 16 + i] * inv);
                a2[i]     = hp[k0 + i];
                a2[i + 8] = hp[k0 + 16 + i];
            }
            acc = __builtin_amdgcn_wmma_f32_16x16x32_bf16(false, a1, false, bl[ks],
                                                          (short)0, acc, false, false);
            acc = __builtin_amdgcn_wmma_f32_16x16x32_bf16(false, a2, false, br[ks],
                                                          (short)0, acc, false, false);
        }

        // epilogue: D-matrix VGPR r holds row (r + hi*8), column o
#pragma unroll
        for (int r = 0; r < 8; r++) {
            float v = acc[r] + bb;
            if (relu) v = fmaxf(v, 0.0f);
            int row = strip * 16 + hi * 8 + r;
            out[(size_t)row * DIM + o] = (__bf16)v;
        }
    }
}

// -------------------------------------------------------- head: 128->2 + softmax
__global__ void k_head(const __bf16* __restrict__ h, const float* __restrict__ Wlin,
                       const float* __restrict__ blin, float* __restrict__ preds, int N) {
    __shared__ float w[2 * DIM];
    if (threadIdx.x < 2 * DIM) w[threadIdx.x] = Wlin[threadIdx.x];
    __syncthreads();
    int n = blockIdx.x * blockDim.x + threadIdx.x;
    if (n >= N) return;
    const __bf16* hp = h + (size_t)n * DIM;
    float d0 = blin[0], d1 = blin[1];
#pragma unroll 8
    for (int k = 0; k < DIM; k++) {
        float v = (float)hp[k];
        d0 += v * w[k];
        d1 += v * w[DIM + k];
    }
    float mx = fmaxf(d0, d1);
    float e0 = __expf(d0 - mx), e1 = __expf(d1 - mx);
    float s = e0 + e1;
    preds[n * 2 + 0] = e0 / s;
    preds[n * 2 + 1] = e1 / s;
}

// ------------------------------------------- CE loss on already-softmaxed preds
__global__ void k_loss(const float* __restrict__ preds, const int* __restrict__ batch,
                       const int* __restrict__ labels, float* __restrict__ loss, int B) {
    __shared__ float red[256];
    int i = blockIdx.x * blockDim.x + threadIdx.x;
    float v = 0.0f;
    if (i < B) {
        int n = batch[i];
        int y = labels[n];
        float p0 = preds[n * 2], p1 = preds[n * 2 + 1];
        float mx = fmaxf(p0, p1);
        float lse = mx + __logf(__expf(p0 - mx) + __expf(p1 - mx));
        float py = y ? p1 : p0;
        v = -(py - lse);
    }
    red[threadIdx.x] = v;
    __syncthreads();
    for (int s = 128; s > 0; s >>= 1) {
        if (threadIdx.x < s) red[threadIdx.x] += red[threadIdx.x + s];
        __syncthreads();
    }
    if (threadIdx.x == 0) atomicAdd(loss, red[0] / (float)B);
}

// ---------------------------------------------------------------- launch chain
extern "C" void kernel_launch(void* const* d_in, const int* in_sizes, int n_in,
                              void* d_out, int out_size, void* d_ws, size_t ws_size,
                              hipStream_t stream) {
    const float* x      = (const float*)d_in[0];
    const int*   ei     = (const int*)d_in[1];
    const int*   batch  = (const int*)d_in[2];
    const int*   labels = (const int*)d_in[3];
    const float* W1l = (const float*)d_in[4];
    const float* b1  = (const float*)d_in[5];
    const float* W1r = (const float*)d_in[6];
    const float* W2l = (const float*)d_in[7];
    const float* b2  = (const float*)d_in[8];
    const float* W2r = (const float*)d_in[9];
    const float* W3l = (const float*)d_in[10];
    const float* b3  = (const float*)d_in[11];
    const float* W3r = (const float*)d_in[12];
    const float* Wlin = (const float*)d_in[13];
    const float* blin = (const float*)d_in[14];

    const int N = in_sizes[0] / DIM;   // 50000
    const int E = in_sizes[1] / 2;     // 800000
    const int B = in_sizes[2];         // 25000
    const int* src = ei;
    const int* dst = ei + E;

    // workspace carve-up (all recomputed every call; no cross-call state)
    char* ws = (char*)d_ws;
    size_t off = 0;
    auto carve = [&](size_t bytes) -> char* {
        char* p = ws + off;
        off += (bytes + 255) & ~(size_t)255;
        return p;
    };
    float*  invc  = (float*)carve((size_t)N * 4);
    float*  agg   = (float*)carve((size_t)N * DIM * 4);
    __bf16* hA    = (__bf16*)carve((size_t)N * DIM * 2);
    __bf16* hB    = (__bf16*)carve((size_t)N * DIM * 2);
    float*  preds = (float*)carve((size_t)N * 2 * 4);
    __bf16* wb[6];
    for (int i = 0; i < 6; i++) wb[i] = (__bf16*)carve((size_t)DIM * DIM * 2);
    (void)ws_size; (void)n_in; (void)out_size;

    const int T  = 256;
    const int ND = N * DIM;
    const int WN = DIM * DIM;                    // weight elements
    const int gemmBlocks = (N / 16 + 3) / 4;     // 4 strips of 16 rows per block

    // degree -> inverse counts (shared by all layers)
    k_zero_f32<<<(N + T - 1) / T, T, 0, stream>>>(invc, N);
    k_count<<<(E + T - 1) / T, T, 0, stream>>>(dst, invc, E);
    k_invcnt<<<(N + T - 1) / T, T, 0, stream>>>(invc, N);

    // pre-convert weights to bf16 (once per call)
    const float* wsrc[6] = { W1l, W1r, W2l, W2r, W3l, W3r };
    for (int i = 0; i < 6; i++)
        k_f32_to_bf16<<<(WN + T - 1) / T, T, 0, stream>>>(wsrc[i], wb[i], WN);

    // x -> bf16 resident features
    k_f32_to_bf16<<<(ND + T - 1) / T, T, 0, stream>>>(x, hA, ND);

    const int scatterBlocks = (int)(((size_t)E * 32 + T - 1) / T);

    // layer 1: hA -> hB (relu)
    k_zero_f32<<<(ND + T - 1) / T, T, 0, stream>>>(agg, ND);
    k_scatter<<<scatterBlocks, T, 0, stream>>>(hA, src, dst, agg, E);
    k_sage_gemm<<<gemmBlocks, T, 0, stream>>>(agg, invc, hA, wb[0], wb[1], b1, hB, 1, N);

    // layer 2: hB -> hA (relu)
    k_zero_f32<<<(ND + T - 1) / T, T, 0, stream>>>(agg, ND);
    k_scatter<<<scatterBlocks, T, 0, stream>>>(hB, src, dst, agg, E);
    k_sage_gemm<<<gemmBlocks, T, 0, stream>>>(agg, invc, hB, wb[2], wb[3], b2, hA, 1, N);

    // layer 3: hA -> hB (no relu)
    k_zero_f32<<<(ND + T - 1) / T, T, 0, stream>>>(agg, ND);
    k_scatter<<<scatterBlocks, T, 0, stream>>>(hA, src, dst, agg, E);
    k_sage_gemm<<<gemmBlocks, T, 0, stream>>>(agg, invc, hA, wb[4], wb[5], b3, hB, 0, N);

    // head + loss
    k_head<<<(N + T - 1) / T, T, 0, stream>>>(hB, Wlin, blin, preds, N);
    k_zero_f32<<<1, 32, 0, stream>>>((float*)d_out, 1);
    k_loss<<<(B + T - 1) / T, T, 0, stream>>>(preds, batch, labels, (float*)d_out, B);
}
